// DenseFlashBlock_84267258347729
// MI455X (gfx1250) — compile-verified, matrix operand order checked by reading, not run on hardware
//
#include <hip/hip_runtime.h>
#include <hip/hip_bf16.h>

// ---------------------------------------------------------------------------
// GPT-2 block on MI455X (gfx1250): bf16 WMMA, f32 accumulate,
// async global->LDS staging (ASYNCcnt) with double buffering.
// ---------------------------------------------------------------------------

#define Bz   2
#define Sz   2048
#define Dz   1024
#define Hz   16
#define DHz  64
#define DFFz 4096
#define Mz   (Bz * Sz)          // 4096 rows of activations

typedef unsigned short u16t;
typedef __attribute__((ext_vector_type(16))) __bf16       v16bf;
typedef __attribute__((ext_vector_type(8)))  float        v8f;
typedef __attribute__((ext_vector_type(8)))  unsigned int v8u;

// exact param types of __builtin_amdgcn_global_load_async_to_lds_b128:
// (v4i addrspace(1)*, v4i addrspace(3)*, imm int, imm int)
typedef int v4i_t __attribute__((vector_size(16)));
typedef __attribute__((address_space(1))) v4i_t* gv4p;
typedef __attribute__((address_space(3))) v4i_t* lv4p;

#if defined(__HIP_DEVICE_COMPILE__) && \
    __has_builtin(__builtin_amdgcn_global_load_async_to_lds_b128)
#define HAS_ASYNC_LDS 1
#else
#define HAS_ASYNC_LDS 0
#endif

// ---------------------------------------------------------------- helpers ---
__device__ __forceinline__ u16t f2bf(float f) {
  unsigned int u = __float_as_uint(f);
  u += 0x7fffu + ((u >> 16) & 1u);          // round-to-nearest-even
  return (u16t)(u >> 16);
}

// 16-byte global -> LDS copy. Async DMA path on CDNA5 (tracked by ASYNCcnt);
// synchronous VGPR bounce as fallback.
__device__ __forceinline__ void async_cp16(const void* g, void* l) {
#if HAS_ASYNC_LDS
  __builtin_amdgcn_global_load_async_to_lds_b128((gv4p)g, (lv4p)l, 0, 0);
#else
  *(uint4*)l = *(const uint4*)g;
#endif
}

__device__ __forceinline__ void wait_async() {
#if defined(__HIP_DEVICE_COMPILE__)
#if __has_builtin(__builtin_amdgcn_s_wait_asynccnt)
  __builtin_amdgcn_s_wait_asynccnt(0);
#else
  asm volatile("s_wait_asynccnt 0" ::: "memory");
#endif
#endif
}

__device__ __forceinline__ v8f wmma_bf16(v8u a, v8u b, v8f c) {
  return __builtin_amdgcn_wmma_f32_16x16x32_bf16(
      false, __builtin_bit_cast(v16bf, a),
      false, __builtin_bit_cast(v16bf, b),
      (short)0, c, false, false);
}

// Load a 16x32 bf16 fragment (A-layout; B uses the same with N in place of M).
// Per ISA: lane m = lane&15, half = lane>>4; dword j holds the K-pair at
// K = (j>>2)*16 + half*8 + (j&3)*2.
__device__ __forceinline__ v8u load_frag16(const u16t* base, int ld) {
  const int lane = threadIdx.x & 31;
  const u16t* p = base + (lane & 15) * ld + ((lane >> 4) << 3);
  v8u r;
#pragma unroll
  for (int j = 0; j < 8; ++j) {
    const int kb = ((j >> 2) << 4) + ((j & 3) << 1);
    r[j] = *(const unsigned int*)(p + kb);
  }
  return r;
}

__device__ __forceinline__ float gelu_tanh(float x) {
  const float c = 0.7978845608028654f;
  float t = tanhf(c * (x + 0.044715f * x * x * x));
  return 0.5f * x * (1.0f + t);
}

// ----------------------------------------------- transpose + cast kernel ---
// out[N][K] (bf16) = transpose(in[K][N]) (f32). 32x32 tiles via LDS.
__global__ __launch_bounds__(256) void castT_bf16_kernel(
    const float* __restrict__ in, u16t* __restrict__ out, int Kd, int Nd) {
  __shared__ float tile[32][33];
  const int tx = threadIdx.x & 31;
  const int ty = threadIdx.x >> 5;          // 0..7
  const int n0 = blockIdx.x * 32;
  const int k0 = blockIdx.y * 32;
#pragma unroll
  for (int i = 0; i < 32; i += 8)
    tile[ty + i][tx] = in[(size_t)(k0 + ty + i) * Nd + n0 + tx];
  __syncthreads();
#pragma unroll
  for (int i = 0; i < 32; i += 8) {
    const int n = ty + i;
    out[(size_t)(n0 + n) * Kd + k0 + tx] = f2bf(tile[tx][n]);
  }
}

// ------------------------------------------------------------- layernorm ---
// One row (D=1024) per 256-thread block; bf16 output.
__global__ __launch_bounds__(256) void ln_bf16_kernel(
    const float* __restrict__ x, const float* __restrict__ g,
    const float* __restrict__ bta, u16t* __restrict__ out) {
  __shared__ float s1[256], s2[256];
  const int row = blockIdx.x;
  const int t = threadIdx.x;
  const float* xr = x + (size_t)row * Dz;
  float4 xv = *(const float4*)(xr + t * 4);
  s1[t] = xv.x + xv.y + xv.z + xv.w;
  s2[t] = xv.x * xv.x + xv.y * xv.y + xv.z * xv.z + xv.w * xv.w;
  __syncthreads();
  for (int o = 128; o > 0; o >>= 1) {
    if (t < o) { s1[t] += s1[t + o]; s2[t] += s2[t + o]; }
    __syncthreads();
  }
  const float mu = s1[0] * (1.0f / Dz);
  const float var = s2[0] * (1.0f / Dz) - mu * mu;
  const float rs = rsqrtf(var + 1e-5f);
  const int c = t * 4;
  float y0 = (xv.x - mu) * rs * g[c + 0] + bta[c + 0];
  float y1 = (xv.y - mu) * rs * g[c + 1] + bta[c + 1];
  float y2 = (xv.z - mu) * rs * g[c + 2] + bta[c + 2];
  float y3 = (xv.w - mu) * rs * g[c + 3] + bta[c + 3];
  unsigned int lo = (unsigned int)f2bf(y0) | ((unsigned int)f2bf(y1) << 16);
  unsigned int hi = (unsigned int)f2bf(y2) | ((unsigned int)f2bf(y3) << 16);
  *(uint2*)(out + (size_t)row * Dz + c) = uint2{lo, hi};
}

// ----------------------------------------------------------- WMMA GEMM -----
// C[M,N] = A[M,K](bf16,row-major) * Bt[N,K](bf16, pre-transposed) + bias.
// Block tile 128x128, BK=32, 8 waves (4x2), 32x64 per wave, 2x4 wmma frags.
// Double-buffered async global->LDS staging.
// FUSE: 0 = none, 1 = tanh-GELU, 2 = residual add. OUTBF: bf16 vs f32 out.
template <int FUSE, int OUTBF>
__global__ __launch_bounds__(256) void gemm_bf16_kernel(
    const u16t* __restrict__ A, const u16t* __restrict__ Bt,
    const float* __restrict__ bias, const float* __restrict__ resid,
    void* __restrict__ Cout, int M, int N, int K) {
  __shared__ u16t As[2][128][32];
  __shared__ u16t Bs[2][128][32];

  const int t = threadIdx.x;
  const int col0 = blockIdx.x * 128;
  const int row0 = blockIdx.y * 128;

  const int wave = t >> 5;
  const int lane = t & 31;
  const int m0 = (wave & 3) * 32;   // wave row offset in tile
  const int n0 = (wave >> 2) * 64;  // wave col offset in tile

  v8f acc[2][4];
#pragma unroll
  for (int i = 0; i < 2; ++i)
#pragma unroll
    for (int j = 0; j < 4; ++j)
#pragma unroll
      for (int r = 0; r < 8; ++r) acc[i][j][r] = 0.0f;

  // staging: thread -> (row 0..127, 16-elem k-half); 2x16B per matrix.
  const int s_row = t >> 1;
  const int s_kh  = (t & 1) << 4;

  auto stage = [&](int buf, int kt) {
    const int k0 = kt << 5;
    const u16t* ga = A + (size_t)(row0 + s_row) * K + k0 + s_kh;
    async_cp16(ga,     &As[buf][s_row][s_kh]);
    async_cp16(ga + 8, &As[buf][s_row][s_kh + 8]);
    const u16t* gb = Bt + (size_t)(col0 + s_row) * K + k0 + s_kh;
    async_cp16(gb,     &Bs[buf][s_row][s_kh]);
    async_cp16(gb + 8, &Bs[buf][s_row][s_kh + 8]);
  };

  const int kTiles = K >> 5;
  stage(0, 0);
  for (int kt = 0; kt < kTiles; ++kt) {
    wait_async();                 // this buffer's DMA done (per wave)
    __syncthreads();              // ... and for every wave in the group
    if (kt + 1 < kTiles) stage((kt + 1) & 1, kt + 1);  // overlap with compute
    const int buf = kt & 1;

    v8u fa[2], fb[4];
#pragma unroll
    for (int i = 0; i < 2; ++i) fa[i] = load_frag16(&As[buf][m0 + 16 * i][0], 32);
#pragma unroll
    for (int j = 0; j < 4; ++j) fb[j] = load_frag16(&Bs[buf][n0 + 16 * j][0], 32);
#pragma unroll
    for (int i = 0; i < 2; ++i)
#pragma unroll
      for (int j = 0; j < 4; ++j) acc[i][j] = wmma_bf16(fa[i], fb[j], acc[i][j]);
  }

  // Epilogue. C layout: VGPR r -> m = r + 8*(lane>>4), n = lane&15.
  const int cn = lane & 15;
  const int rb = (lane >> 4) << 3;
#pragma unroll
  for (int i = 0; i < 2; ++i) {
#pragma unroll
    for (int j = 0; j < 4; ++j) {
      const int gn = col0 + n0 + 16 * j + cn;
      const float bv = bias[gn];
      const int gmb = row0 + m0 + 16 * i + rb;
#pragma unroll
      for (int r = 0; r < 8; ++r) {
        float v = acc[i][j][r] + bv;
        const size_t idx = (size_t)(gmb + r) * N + gn;
        if (FUSE == 1) v = gelu_tanh(v);
        if (FUSE == 2) v += resid[idx];
        if (OUTBF) ((u16t*)Cout)[idx] = f2bf(v);
        else       ((float*)Cout)[idx] = v;
      }
    }
  }
}

// ----------------------------------------------------- flash attention -----
// qkv: [B,S,3D] bf16 (q|k|v). One block = (b, h, 128 q rows); 8 waves, each
// owns a 16-row Q strip. KV tiles of 32 keys; online softmax; causal mask.
__global__ __launch_bounds__(256) void flash_attn_kernel(
    const u16t* __restrict__ qkv, u16t* __restrict__ Y) {
  __shared__ u16t Ks[32][64];      // K tile  [key][d]   (async staged)
  __shared__ u16t Vt[64][32];      // V tile transposed [d][key]
  __shared__ u16t Ps[8][16][32];   // per-wave P scratch [m][key]

  const int blk = blockIdx.x;
  const int qt = blk & 15;                  // q tile of 128 rows
  const int bh = blk >> 4;
  const int h = bh % Hz;
  const int b = bh / Hz;

  const int t = threadIdx.x;
  const int wave = t >> 5;
  const int lane = t & 31;
  const int half = lane >> 4;
  const int ln = lane & 15;
  const int q0 = qt * 128 + wave * 16;

  const size_t rowStride = 3 * (size_t)Dz;
  const u16t* qkvB = qkv + (size_t)b * Sz * rowStride;

  // --- load Q fragments (16 rows x 64 d) straight from global (bf16) ------
  v8u fq[2];
  {
    const u16t* qb = qkvB + (size_t)(q0 + ln) * rowStride + h * DHz;
#pragma unroll
    for (int f = 0; f < 2; ++f)
#pragma unroll
      for (int j = 0; j < 8; ++j) {
        const int kb = f * 32 + ((j >> 2) << 4) + (half << 3) + ((j & 3) << 1);
        fq[f][j] = *(const unsigned int*)(qb + kb);
      }
  }

  v8f accv[4];
  float rmax[8], rsum[8];
#pragma unroll
  for (int c = 0; c < 4; ++c)
#pragma unroll
    for (int r = 0; r < 8; ++r) accv[c][r] = 0.0f;
#pragma unroll
  for (int r = 0; r < 8; ++r) { rmax[r] = -3.0e38f; rsum[r] = 0.0f; }

  // staging mapping: thread -> (key 0..31, 8-elem d-group)
  const int s_key = t >> 3;
  const int s_dg  = (t & 7) << 3;

  const int nTiles = (qt + 1) * 4;          // causal upper bound for block
  for (int kt = 0; kt < nTiles; ++kt) {
    const int k0 = kt * 32;
    // ---- stage K (async DMA) and V^T (manual transpose) -----------------
    {
      const u16t* kp = qkvB + (size_t)(k0 + s_key) * rowStride + Dz + h * DHz + s_dg;
      async_cp16(kp, &Ks[s_key][s_dg]);
      const u16t* vp = qkvB + (size_t)(k0 + s_key) * rowStride + 2 * Dz + h * DHz + s_dg;
      uint4 vv = *(const uint4*)vp;
      union { uint4 u; u16t s[8]; } uv; uv.u = vv;
#pragma unroll
      for (int i = 0; i < 8; ++i) Vt[s_dg + i][s_key] = uv.s[i];
    }
    wait_async();
    __syncthreads();

    // ---- scores: S = Q @ K^T (two 16-key chunks), scale + mask ----------
    float sc[2][8];
#pragma unroll
    for (int ch = 0; ch < 2; ++ch) {
      v8u fk0 = load_frag16(&Ks[ch * 16][0], 64);   // d 0..31
      v8u fk1 = load_frag16(&Ks[ch * 16][32], 64);  // d 32..63
      v8f s;
#pragma unroll
      for (int r = 0; r < 8; ++r) s[r] = 0.0f;
      s = wmma_bf16(fq[0], fk0, s);
      s = wmma_bf16(fq[1], fk1, s);
      const int kg = k0 + ch * 16 + ln;             // global key (lane col)
#pragma unroll
      for (int r = 0; r < 8; ++r) {
        const int qr = q0 + r + (half << 3);
        float v = s[r] * 0.125f;                    // 1/sqrt(64)
        sc[ch][r] = (kg > qr) ? -3.0e38f : v;
      }
    }

    // ---- online softmax update (16-lane row reductions) -----------------
#pragma unroll
    for (int r = 0; r < 8; ++r) {
      float mloc = fmaxf(sc[0][r], sc[1][r]);
      mloc = fmaxf(mloc, __shfl_xor(mloc, 1, 32));
      mloc = fmaxf(mloc, __shfl_xor(mloc, 2, 32));
      mloc = fmaxf(mloc, __shfl_xor(mloc, 4, 32));
      mloc = fmaxf(mloc, __shfl_xor(mloc, 8, 32));
      const float mnew = fmaxf(rmax[r], mloc);
      const float alpha = __expf(rmax[r] - mnew);
      rmax[r] = mnew;
      const float p0 = __expf(sc[0][r] - mnew);
      const float p1 = __expf(sc[1][r] - mnew);
      float ls = p0 + p1;
      ls += __shfl_xor(ls, 1, 32);
      ls += __shfl_xor(ls, 2, 32);
      ls += __shfl_xor(ls, 4, 32);
      ls += __shfl_xor(ls, 8, 32);
      rsum[r] = rsum[r] * alpha + ls;
#pragma unroll
      for (int c = 0; c < 4; ++c) accv[c][r] *= alpha;
      const int m = r + (half << 3);
      Ps[wave][m][ln] = f2bf(p0);
      Ps[wave][m][16 + ln] = f2bf(p1);
    }
    __syncthreads();

    // ---- O += P @ V ------------------------------------------------------
    v8u fp = load_frag16(&Ps[wave][0][0], 32);
#pragma unroll
    for (int c = 0; c < 4; ++c) {
      v8u fv = load_frag16(&Vt[c * 16][0], 32);
      accv[c] = wmma_bf16(fp, fv, accv[c]);
    }
    __syncthreads();                                // tiles reused next iter
  }

  // ---- write Y (bf16) ----------------------------------------------------
  float inv[8];
#pragma unroll
  for (int r = 0; r < 8; ++r) inv[r] = 1.0f / rsum[r];
#pragma unroll
  for (int c = 0; c < 4; ++c)
#pragma unroll
    for (int r = 0; r < 8; ++r) {
      const int m = r + (half << 3);
      const size_t idx = ((size_t)b * Sz + q0 + m) * Dz + h * DHz + c * 16 + ln;
      Y[idx] = f2bf(accv[c][r] * inv[r]);
    }
}

// ------------------------------------------------------------ launch -------
extern "C" void kernel_launch(void* const* d_in, const int* in_sizes, int n_in,
                              void* d_out, int out_size, void* d_ws, size_t ws_size,
                              hipStream_t stream) {
  const float* hidden = (const float*)d_in[0];
  const float* ln1_g  = (const float*)d_in[1];
  const float* ln1_b  = (const float*)d_in[2];
  const float* w_attn = (const float*)d_in[3];
  const float* b_attn = (const float*)d_in[4];
  const float* w_proj = (const float*)d_in[5];
  const float* b_proj = (const float*)d_in[6];
  const float* ln2_g  = (const float*)d_in[7];
  const float* ln2_b  = (const float*)d_in[8];
  const float* w_fc   = (const float*)d_in[9];
  const float* b_fc   = (const float*)d_in[10];
  const float* w_fc2  = (const float*)d_in[11];
  const float* b_fc2  = (const float*)d_in[12];

  char* p = (char*)d_ws;
  auto alloc = [&](size_t bytes) {
    char* r = p;
    p += (bytes + 255) & ~(size_t)255;
    return r;
  };
  u16t*  xa    = (u16t*)alloc((size_t)Mz * Dz * 2);          // ln1 out
  u16t*  wqkvT = (u16t*)alloc((size_t)Dz * 3 * Dz * 2);      // [3D][D]
  u16t*  qkv   = (u16t*)alloc((size_t)Mz * 3 * Dz * 2);
  u16t*  ybf   = (u16t*)alloc((size_t)Mz * Dz * 2);          // attn out
  u16t*  wprjT = (u16t*)alloc((size_t)Dz * Dz * 2);          // [D][D]
  float* hid2  = (float*)alloc((size_t)Mz * Dz * 4);         // post-attn resid
  u16t*  zb    = (u16t*)alloc((size_t)Mz * Dz * 2);          // ln2 out
  u16t*  wfcT  = (u16t*)alloc((size_t)Dz * DFFz * 2);        // [DFF][D]
  u16t*  h1    = (u16t*)alloc((size_t)Mz * DFFz * 2);        // gelu(fc1)
  u16t*  wfc2T = (u16t*)alloc((size_t)DFFz * Dz * 2);        // [D][DFF]

  // transpose-cast weights once: Bt[N][K] bf16 from W[K][N] f32
  auto castT = [&](const float* src, u16t* dst, int Kd, int Nd) {
    castT_bf16_kernel<<<dim3(Nd / 32, Kd / 32), 256, 0, stream>>>(src, dst, Kd, Nd);
  };
  castT(w_attn, wqkvT, Dz, 3 * Dz);
  castT(w_proj, wprjT, Dz, Dz);
  castT(w_fc,   wfcT,  Dz, DFFz);
  castT(w_fc2,  wfc2T, DFFz, Dz);

  // LN1
  ln_bf16_kernel<<<Mz, 256, 0, stream>>>(hidden, ln1_g, ln1_b, xa);

  // QKV = x @ w_attn + b_attn  (bf16 out)
  gemm_bf16_kernel<0, 1><<<dim3(3 * Dz / 128, Mz / 128), 256, 0, stream>>>(
      xa, wqkvT, b_attn, nullptr, qkv, Mz, 3 * Dz, Dz);

  // Flash attention  (bf16 out)
  flash_attn_kernel<<<Bz * Hz * (Sz / 128), 256, 0, stream>>>(qkv, ybf);

  // hid2 = hidden + Y @ w_proj + b_proj  (f32)
  gemm_bf16_kernel<2, 0><<<dim3(Dz / 128, Mz / 128), 256, 0, stream>>>(
      ybf, wprjT, b_proj, hidden, hid2, Mz, Dz, Dz);

  // LN2
  ln_bf16_kernel<<<Mz, 256, 0, stream>>>(hid2, ln2_g, ln2_b, zb);

  // h1 = gelu(z @ w_fc + b_fc)  (bf16)
  gemm_bf16_kernel<1, 1><<<dim3(DFFz / 128, Mz / 128), 256, 0, stream>>>(
      zb, wfcT, b_fc, nullptr, h1, Mz, DFFz, Dz);

  // out = hid2 + h1 @ w_fc2 + b_fc2  (f32)
  gemm_bf16_kernel<2, 0><<<dim3(Dz / 128, Mz / 128), 256, 0, stream>>>(
      h1, wfc2T, b_fc2, hid2, (float*)d_out, Mz, Dz, DFFz);
}
